// MultiHeadSelfAttention_76630806495342
// MI455X (gfx1250) — matrix operand order repaired and checked by reading, not compiled, not run on hardware
//
#include <hip/hip_runtime.h>
#include <hip/hip_bf16.h>

// ---------------------------------------------------------------------------
// AutoInt multi-head self-attention, fused, bf16 WMMA (v_wmma_f32_16x16x32_bf16)
// B=16384 batches, F=40 fields (padded to 48), D=64, D_OUT=64.
// One wave32 per batch, 2 batches per 64-thread block.
// ---------------------------------------------------------------------------

typedef __bf16 bf16;
typedef bf16  v16bf __attribute__((ext_vector_type(16)));
typedef bf16  v8bf  __attribute__((ext_vector_type(8)));
typedef bf16  v4bf  __attribute__((ext_vector_type(4)));
typedef float v8f   __attribute__((ext_vector_type(8)));
typedef float v4f   __attribute__((ext_vector_type(4)));

#define NB     16384   // batches
#define NF     40      // fields
#define FPAD   48      // fields padded to multiple of 16
#define DD     64      // embedding dim (= K of projections)
#define EE     64      // output embedding dim
#define LS     72      // LDS row stride (elems); 144B rows keep b128 16B-aligned

#define PER_WAVE_ELEMS (3 * FPAD * LS + EE * LS)   // Xb + Kb + QS + Vt

// A-fragment (16x32 bf16, lane r = row): elems j = k = kbase + h*16 + j.
// Two 16B LDS loads.
__device__ __forceinline__ v16bf ld_a_frag(const bf16* row, int kbase, int h) {
  v8bf lo = *(const v8bf*)(row + kbase + h * 16);
  v8bf hi = *(const v8bf*)(row + kbase + h * 16 + 8);
  return __builtin_shufflevector(lo, hi, 0,1,2,3,4,5,6,7,8,9,10,11,12,13,14,15);
}

// B-fragment (32x16 bf16, lane c = col): elems j = k = kbase + h*16 + j,
// read from a row-major-in-k buffer (row = column of B). Two 16B LDS loads.
__device__ __forceinline__ v16bf ld_b_frag(const bf16* row, int kbase, int h) {
  v8bf lo = *(const v8bf*)(row + kbase + h * 16);
  v8bf hi = *(const v8bf*)(row + kbase + h * 16 + 8);
  return __builtin_shufflevector(lo, hi, 0,1,2,3,4,5,6,7,8,9,10,11,12,13,14,15);
}

// Weight B-fragment gathered from global fp32 W[d][e] (row-major, e fastest):
// lane (h,c) needs W[kt*32 + h*16 + j][nt*16 + c] for j=0..15.
__device__ __forceinline__ v16bf ld_wb_frag(const float* __restrict__ W,
                                            int kt, int nt, int h, int c) {
  const float* p = W + (size_t)(kt * 32 + h * 16) * EE + nt * 16 + c;
  v16bf w;
#pragma unroll
  for (int j = 0; j < 16; ++j) w[j] = (bf16)p[(size_t)j * EE];
  return w;
}

__device__ __forceinline__ v8f wmma_bf16(v16bf a, v16bf b, v8f c) {
  return __builtin_amdgcn_wmma_f32_16x16x32_bf16(false, a, false, b,
                                                 (short)0, c, false, false);
}

__global__ void __launch_bounds__(64)
attn_autoint_kernel(const float* __restrict__ X,
                    const float* __restrict__ Wq,
                    const float* __restrict__ Wk,
                    const float* __restrict__ Wv,
                    const float* __restrict__ Wr,
                    float* __restrict__ Out) {
  __shared__ __align__(16) bf16 smem[2 * PER_WAVE_ELEMS];

  const int lane = threadIdx.x & 31;
  const int wave = threadIdx.x >> 5;
  const int h = lane >> 4;        // half-wave select
  const int r = lane & 15;        // row (A) / col (B,C) within tile
  const int batch = blockIdx.x * 2 + wave;

  bf16* base = smem + wave * PER_WAVE_ELEMS;
  bf16* Xb = base;                  // [FPAD][LS] bf16 X, rows 40..47 zero
  bf16* Kb = base + FPAD * LS;      // [FPAD][LS] K projection, row-major [f][e]
  bf16* QS = base + 2 * FPAD * LS;  // [FPAD][LS] Q, later reused for S [q][k]
  bf16* Vt = base + 3 * FPAD * LS;  // [EE][LS]  V transposed: Vt[e][f]

  const float* Xg = X + (size_t)batch * NF * DD;
  float*       Og = Out + (size_t)batch * NF * EE;

  // ---- Phase 1: X -> bf16 LDS (coalesced float4 loads), zero pad regions ----
#pragma unroll
  for (int t = 0; t < 20; ++t) {
    int idx = lane + t * 32;           // 0..639 float4 chunks
    int row = idx >> 4;
    int c4  = (idx & 15) * 4;
    v4f xv = *(const v4f*)(Xg + (size_t)row * DD + c4);
    v4bf pk;
    pk[0] = (bf16)xv.x; pk[1] = (bf16)xv.y;
    pk[2] = (bf16)xv.z; pk[3] = (bf16)xv.w;
    *(v4bf*)(Xb + row * LS + c4) = pk;
  }
  {
    v8bf z = {};
    // zero X rows 40..47 (cols 0..63): each lane does 16 elems
    int zr = 40 + (lane >> 2);
    int zc = (lane & 3) * 16;
    *(v8bf*)(Xb + zr * LS + zc) = z;
    *(v8bf*)(Xb + zr * LS + zc + 8) = z;
    // zero Vt[:, 48..63] so S@V k-tile 1 reads clean zeros
#pragma unroll
    for (int t = 0; t < 2; ++t) {
      int row = lane + t * 32;
      *(v8bf*)(Vt + row * LS + 48) = z;
      *(v8bf*)(Vt + row * LS + 56) = z;
    }
  }

  // ---- Phase 2: projections K, V, Q (X[48x64] @ W[64x64], bf16 WMMA) ----
#pragma unroll
  for (int proj = 0; proj < 3; ++proj) {
    const float* W = (proj == 0) ? Wk : (proj == 1) ? Wv : Wq;
    for (int nt = 0; nt < 4; ++nt) {
      v16bf b0 = ld_wb_frag(W, 0, nt, h, r);
      v16bf b1 = ld_wb_frag(W, 1, nt, h, r);
#pragma unroll
      for (int mt = 0; mt < 3; ++mt) {
        const bf16* xrow = Xb + (mt * 16 + r) * LS;
        v16bf a0 = ld_a_frag(xrow, 0, h);
        v16bf a1 = ld_a_frag(xrow, 32, h);
        v8f acc = {};
        acc = wmma_bf16(a0, b0, acc);
        acc = wmma_bf16(a1, b1, acc);
        if (proj == 1) {
          // V: store transposed Vt[e][f], one b128 per tile
          v8bf pk;
#pragma unroll
          for (int i = 0; i < 8; ++i) pk[i] = (bf16)acc[i];
          *(v8bf*)(Vt + (nt * 16 + r) * LS + mt * 16 + h * 8) = pk;
        } else {
          // K / Q: store row-major [f][e] (needed as A of scores / B of Q^T)
          bf16* dst = (proj == 0) ? Kb : QS;
#pragma unroll
          for (int i = 0; i < 8; ++i)
            dst[(mt * 16 + h * 8 + i) * LS + nt * 16 + r] = (bf16)acc[i];
        }
      }
    }
  }

  // ---- Phase 3: S^T = K @ Q^T, in-register softmax over k, store S[q][k] ----
  for (int nt = 0; nt < 3; ++nt) {           // q tiles
    // B = Q^T: lane (h,c) reads Q row (nt*16+c), contiguous e
    const bf16* qrow = QS + (nt * 16 + r) * LS;
    v16bf qb0 = ld_b_frag(qrow, 0, h);
    v16bf qb1 = ld_b_frag(qrow, 32, h);
    v8f sc[3];
#pragma unroll
    for (int mt = 0; mt < 3; ++mt) {         // k tiles
      const bf16* krow = Kb + (mt * 16 + r) * LS;
      v16bf ka0 = ld_a_frag(krow, 0, h);
      v16bf ka1 = ld_a_frag(krow, 32, h);
      v8f acc = {};
      acc = wmma_bf16(ka0, qb0, acc);
      acc = wmma_bf16(ka1, qb1, acc);
      sc[mt] = acc;
    }
    // softmax over k (spread over 3 accs in-lane + the opposite half-wave)
    float mx = -3.0e38f;
#pragma unroll
    for (int mt = 0; mt < 3; ++mt)
#pragma unroll
      for (int i = 0; i < 8; ++i) {
        bool valid = !(mt == 2 && h == 1);   // k = mt*16+h*8+i >= 40 is pad
        mx = valid ? fmaxf(mx, sc[mt][i]) : mx;
      }
    mx = fmaxf(mx, __shfl_xor(mx, 16, 32));
    float ex[3][8];
    float sum = 0.f;
#pragma unroll
    for (int mt = 0; mt < 3; ++mt)
#pragma unroll
      for (int i = 0; i < 8; ++i) {
        bool valid = !(mt == 2 && h == 1);
        float e = valid ? __expf(sc[mt][i] - mx) : 0.f;
        ex[mt][i] = e;
        sum += e;
      }
    sum += __shfl_xor(sum, 16, 32);
    float rs = 1.0f / sum;
    // store normalized S row-major [q][k] into QS (Q is dead for this q tile)
#pragma unroll
    for (int mt = 0; mt < 3; ++mt) {
      v8bf pk;
#pragma unroll
      for (int i = 0; i < 8; ++i) pk[i] = (bf16)(ex[mt][i] * rs);
      *(v8bf*)(QS + (nt * 16 + r) * LS + mt * 16 + h * 8) = pk;
    }
    if (h == 0) {                            // zero k = 48..63 of this row
      v8bf z = {};
      *(v8bf*)(QS + (nt * 16 + r) * LS + 48) = z;
      *(v8bf*)(QS + (nt * 16 + r) * LS + 56) = z;
    }
  }

  // ---- Phase 4: Out = relu(X@Wr + S@V), fused accumulation ----
  for (int nt = 0; nt < 4; ++nt) {           // e tiles
    v16bf wr0 = ld_wb_frag(Wr, 0, nt, h, r);
    v16bf wr1 = ld_wb_frag(Wr, 1, nt, h, r);
    const bf16* vrow = Vt + (nt * 16 + r) * LS;
    v16bf vb0 = ld_b_frag(vrow, 0, h);
    v16bf vb1 = ld_b_frag(vrow, 32, h);
#pragma unroll
    for (int mt = 0; mt < 3; ++mt) {
      const bf16* xrow = Xb + (mt * 16 + r) * LS;
      v16bf a0 = ld_a_frag(xrow, 0, h);
      v16bf a1 = ld_a_frag(xrow, 32, h);
      v8f acc = {};
      acc = wmma_bf16(a0, wr0, acc);         // residual X @ W_Res
      acc = wmma_bf16(a1, wr1, acc);
      const bf16* srow = QS + (mt * 16 + r) * LS;
      v16bf s0 = ld_a_frag(srow, 0, h);
      v16bf s1 = ld_a_frag(srow, 32, h);
      acc = wmma_bf16(s0, vb0, acc);         // + S @ V
      acc = wmma_bf16(s1, vb1, acc);
#pragma unroll
      for (int i = 0; i < 8; ++i) {
        int f = mt * 16 + h * 8 + i;
        if (f < NF) Og[(size_t)f * EE + nt * 16 + r] = fmaxf(acc[i], 0.f);
      }
    }
  }
}

extern "C" void kernel_launch(void* const* d_in, const int* in_sizes, int n_in,
                              void* d_out, int out_size, void* d_ws, size_t ws_size,
                              hipStream_t stream) {
  (void)in_sizes; (void)n_in; (void)d_ws; (void)ws_size; (void)out_size;
  const float* X  = (const float*)d_in[0];
  const float* Wq = (const float*)d_in[1];
  const float* Wk = (const float*)d_in[2];
  const float* Wv = (const float*)d_in[3];
  const float* Wr = (const float*)d_in[4];
  float* Out = (float*)d_out;
  dim3 grid(NB / 2);   // 2 batches (waves) per block
  dim3 block(64);
  attn_autoint_kernel<<<grid, block, 0, stream>>>(X, Wq, Wk, Wv, Wr, Out);
}
